// MyRnn_86500641341581
// MI455X (gfx1250) — compile-verified, matrix operand order checked by reading, not run on hardware
//
#include <hip/hip_runtime.h>
#include <hip/hip_bf16.h>

typedef __attribute__((ext_vector_type(16))) __bf16 v16bf;
typedef __attribute__((ext_vector_type(8)))  __bf16 v8bf;
typedef __attribute__((ext_vector_type(8)))  float  v8f;

#define EMBED 100
#define EPAD  128
#define SEQ   121
#define UNITS 512
#define BATCH 1024
#define G4    2048   // 4*UNITS

// ---------------------------------------------------------------------------
// Fragment loader for v_wmma_f32_16x16x32_bf16 (A and B share the layout when
// B is stored row-major as [N,K], i.e. pre-transposed weights).
// Lane l: half = l>>4, idx = l&15 (M for A, N for B).
// Elements e=0..7  -> K = kbase + half*8 + e        (contiguous 16 bytes)
// Elements e=8..15 -> K = kbase + half*8 + 16 + e-8 (contiguous 16 bytes)
// ---------------------------------------------------------------------------
__device__ __forceinline__ v16bf load_frag(const __bf16* __restrict__ row,
                                           int kbase, int half) {
  const v8bf lo = *(const v8bf*)(row + kbase + half * 8);
  const v8bf hi = *(const v8bf*)(row + kbase + half * 8 + 16);
  v16bf r;
#pragma unroll
  for (int e = 0; e < 8; ++e) { r[e] = lo[e]; r[e + 8] = hi[e]; }
  return r;
}

__device__ __forceinline__ v8f wmma_bf16(v16bf a, v16bf b, v8f c) {
  return __builtin_amdgcn_wmma_f32_16x16x32_bf16(
      /*neg_a=*/false, a, /*neg_b=*/false, b,
      /*c_mod=*/(short)0, c, /*reuse_a=*/false, /*reuse_b=*/false);
}

__device__ __forceinline__ float sigmoidf_(float x) {
  return 1.0f / (1.0f + __expf(-x));
}

// ---------------------------------------------------------------------------
// Embedding gather + pad to 128 cols, fp32 -> bf16.  X layout: [T][B][EPAD]
// ---------------------------------------------------------------------------
__global__ void embed_pad_kernel(const int* __restrict__ tokens,
                                 const float* __restrict__ emb,
                                 __bf16* __restrict__ X, int total) {
  int i = blockIdx.x * blockDim.x + threadIdx.x;
  if (i >= total) return;
  int e  = i & (EPAD - 1);
  int tb = i >> 7;              // t*BATCH + b
  int b  = tb & (BATCH - 1);
  int t  = tb >> 10;
  float v = 0.0f;
  if (e < EMBED) v = emb[(size_t)tokens[b * SEQ + t] * EMBED + e];
  X[i] = (__bf16)v;
}

// ---------------------------------------------------------------------------
// Transpose + pad + convert: W [K,N] f32 row-major -> Wt [N,Kp] bf16 row-major
// ---------------------------------------------------------------------------
__global__ void pack_wt_kernel(const float* __restrict__ W,
                               __bf16* __restrict__ Wt,
                               int K, int Kp, int N, int total) {
  int i = blockIdx.x * blockDim.x + threadIdx.x;
  if (i >= total) return;
  int kp = i % Kp;
  int n  = i / Kp;
  float v = (kp < K) ? W[(size_t)kp * N + n] : 0.0f;
  Wt[i] = (__bf16)v;
}

__global__ void init_state_kernel(__bf16* h1, __bf16* h2,
                                  float* c1, float* c2, int n) {
  int i = blockIdx.x * blockDim.x + threadIdx.x;
  if (i >= n) return;
  h1[i] = (__bf16)0.0f; h2[i] = (__bf16)0.0f;
  c1[i] = 0.0f;         c2[i] = 0.0f;
}

// ---------------------------------------------------------------------------
// Fused LSTM step:  z = x@W + h@U + b ; gates ; c,h update.
// Wave owns a 32(M)x16(N) slice of h: 2 M-tiles x 4 gate tiles = 8 v8f accums.
// Grid: 1024 waves = 128 blocks x 8 waves.  No divergence -> EXEC all ones.
// ---------------------------------------------------------------------------
__global__ __launch_bounds__(256, 1)
void lstm_step_kernel(const __bf16* __restrict__ xin, int Kx,   // [B,Kx]
                      const __bf16* __restrict__ Wt,            // [G4,Kx]
                      const __bf16* __restrict__ hin,           // [B,UNITS]
                      const __bf16* __restrict__ Ut,            // [G4,UNITS]
                      const float*  __restrict__ bias,          // [G4]
                      float*        __restrict__ c,             // [B,UNITS] in-place
                      __bf16*       __restrict__ hout)          // [B,UNITS]
{
  const int lane = threadIdx.x & 31;
  const int wave = blockIdx.x * (blockDim.x >> 5) + (threadIdx.x >> 5);
  const int tn   = wave & 31;          // N tile 0..31
  const int tm2  = wave >> 5;          // M pair 0..31 (covers 32 rows)
  const int half = lane >> 4;
  const int lr   = lane & 15;

  v8f acc[2][4];
#pragma unroll
  for (int mt = 0; mt < 2; ++mt)
#pragma unroll
    for (int g = 0; g < 4; ++g)
#pragma unroll
      for (int e = 0; e < 8; ++e) acc[mt][g][e] = 0.0f;

  const int m0   = tm2 * 32 + lr;      // A row, first M tile
  const int m1   = m0 + 16;            // A row, second M tile
  const int nrow = tn * 16 + lr;       // B row (pre-transposed weights)

  const __bf16* xr0 = xin + (size_t)m0 * Kx;
  const __bf16* xr1 = xin + (size_t)m1 * Kx;
  for (int k = 0; k < Kx; k += 32) {
    v16bf a0 = load_frag(xr0, k, half);
    v16bf a1 = load_frag(xr1, k, half);
#pragma unroll
    for (int g = 0; g < 4; ++g) {
      v16bf bm = load_frag(Wt + (size_t)(g * UNITS + nrow) * Kx, k, half);
      acc[0][g] = wmma_bf16(a0, bm, acc[0][g]);
      acc[1][g] = wmma_bf16(a1, bm, acc[1][g]);
    }
  }

  const __bf16* hr0 = hin + (size_t)m0 * UNITS;
  const __bf16* hr1 = hin + (size_t)m1 * UNITS;
  for (int k = 0; k < UNITS; k += 32) {
    v16bf a0 = load_frag(hr0, k, half);
    v16bf a1 = load_frag(hr1, k, half);
#pragma unroll
    for (int g = 0; g < 4; ++g) {
      v16bf bm = load_frag(Ut + (size_t)(g * UNITS + nrow) * UNITS, k, half);
      acc[0][g] = wmma_bf16(a0, bm, acc[0][g]);
      acc[1][g] = wmma_bf16(a1, bm, acc[1][g]);
    }
  }

  // Epilogue: lane holds column ncol, rows (tm2*32 + mt*16 + half*8 + v).
  const int ncol = tn * 16 + lr;
  const float bi = bias[0 * UNITS + ncol];
  const float bf = bias[1 * UNITS + ncol];
  const float bg = bias[2 * UNITS + ncol];
  const float bo = bias[3 * UNITS + ncol];

#pragma unroll
  for (int mt = 0; mt < 2; ++mt) {
#pragma unroll
    for (int v = 0; v < 8; ++v) {
      int mrow = tm2 * 32 + mt * 16 + half * 8 + v;
      size_t idx = (size_t)mrow * UNITS + ncol;
      float iv = sigmoidf_(acc[mt][0][v] + bi);
      float fv = sigmoidf_(acc[mt][1][v] + bf);
      float gv = tanhf(acc[mt][2][v] + bg);
      float ov = sigmoidf_(acc[mt][3][v] + bo);
      float cn = fv * c[idx] + iv * gv;
      c[idx] = cn;
      hout[idx] = (__bf16)(ov * tanhf(cn));
    }
  }
}

// ---------------------------------------------------------------------------
// Dense 512 -> 1 + sigmoid
// ---------------------------------------------------------------------------
__global__ void dense_sig_kernel(const __bf16* __restrict__ h2,
                                 const float* __restrict__ Wd,
                                 const float* __restrict__ bd,
                                 float* __restrict__ out) {
  int b = blockIdx.x * blockDim.x + threadIdx.x;
  if (b >= BATCH) return;
  float s = bd[0];
  const __bf16* row = h2 + (size_t)b * UNITS;
#pragma unroll 8
  for (int k = 0; k < UNITS; ++k) s += (float)row[k] * Wd[k];
  out[b] = 1.0f / (1.0f + __expf(-s));
}

// ---------------------------------------------------------------------------
extern "C" void kernel_launch(void* const* d_in, const int* in_sizes, int n_in,
                              void* d_out, int out_size, void* d_ws, size_t ws_size,
                              hipStream_t stream) {
  const int*   tokens = (const int*)  d_in[0];
  const float* emb    = (const float*)d_in[1];
  const float* W1     = (const float*)d_in[2];
  const float* U1     = (const float*)d_in[3];
  const float* b1     = (const float*)d_in[4];
  const float* W2     = (const float*)d_in[5];
  const float* U2     = (const float*)d_in[6];
  const float* b2     = (const float*)d_in[7];
  const float* Wd     = (const float*)d_in[8];
  const float* bd     = (const float*)d_in[9];
  float* out = (float*)d_out;

  // Workspace carve (256B aligned)
  uintptr_t p = (uintptr_t)d_ws;
  auto carve = [&](size_t bytes) -> void* {
    void* r = (void*)p;
    p += (bytes + 255) & ~(size_t)255;
    return r;
  };
  __bf16* X    = (__bf16*)carve((size_t)SEQ * BATCH * EPAD * 2);
  __bf16* Wt1  = (__bf16*)carve((size_t)G4 * EPAD * 2);
  __bf16* Ut1  = (__bf16*)carve((size_t)G4 * UNITS * 2);
  __bf16* Wt2  = (__bf16*)carve((size_t)G4 * UNITS * 2);
  __bf16* Ut2  = (__bf16*)carve((size_t)G4 * UNITS * 2);
  __bf16* h1b[2] = { (__bf16*)carve((size_t)BATCH * UNITS * 2),
                     (__bf16*)carve((size_t)BATCH * UNITS * 2) };
  __bf16* h2b[2] = { (__bf16*)carve((size_t)BATCH * UNITS * 2),
                     (__bf16*)carve((size_t)BATCH * UNITS * 2) };
  float* c1 = (float*)carve((size_t)BATCH * UNITS * 4);
  float* c2 = (float*)carve((size_t)BATCH * UNITS * 4);

  // Prep: embeddings + weight packing + state init
  {
    int total = SEQ * BATCH * EPAD;
    embed_pad_kernel<<<(total + 255) / 256, 256, 0, stream>>>(tokens, emb, X, total);
  }
  {
    int total = G4 * EPAD;
    pack_wt_kernel<<<(total + 255) / 256, 256, 0, stream>>>(W1, Wt1, EMBED, EPAD, G4, total);
  }
  {
    int total = G4 * UNITS;
    pack_wt_kernel<<<(total + 255) / 256, 256, 0, stream>>>(U1, Ut1, UNITS, UNITS, G4, total);
    pack_wt_kernel<<<(total + 255) / 256, 256, 0, stream>>>(W2, Wt2, UNITS, UNITS, G4, total);
    pack_wt_kernel<<<(total + 255) / 256, 256, 0, stream>>>(U2, Ut2, UNITS, UNITS, G4, total);
  }
  {
    int total = BATCH * UNITS;
    init_state_kernel<<<(total + 255) / 256, 256, 0, stream>>>(h1b[0], h2b[0], c1, c2, total);
  }

  // Recurrent scan: 1024 waves per step-kernel = 128 blocks x 256 threads
  const int stepBlocks = (BATCH / 32) * (UNITS / 16) / 8;  // 128
  for (int t = 0; t < SEQ; ++t) {
    int cur = t & 1, nxt = cur ^ 1;
    lstm_step_kernel<<<stepBlocks, 256, 0, stream>>>(
        X + (size_t)t * BATCH * EPAD, EPAD, Wt1, h1b[cur], Ut1, b1, c1, h1b[nxt]);
    lstm_step_kernel<<<stepBlocks, 256, 0, stream>>>(
        h1b[nxt], UNITS, Wt2, h2b[cur], Ut2, b2, c2, h2b[nxt]);
  }

  // Final dense + sigmoid (last write was into index (SEQ-1 & 1)^1 == SEQ & 1)
  dense_sig_kernel<<<(BATCH + 255) / 256, 256, 0, stream>>>(h2b[SEQ & 1], Wd, bd, out);
}
